// LSTM_51505247813850
// MI455X (gfx1250) — matrix operand order repaired and checked by reading, not compiled, not run on hardware
//
#include <hip/hip_runtime.h>

// ---------------------------------------------------------------------------
// LSTM (B=56,T=750,I=188,H=512) + dense head, CDNA5 / gfx1250
// wave32 WMMA f16, TDM staging of all step operands, software-pipelined
// ds_load -> v_wmma inner loops.
// ---------------------------------------------------------------------------

typedef __attribute__((ext_vector_type(16))) _Float16 v16h;
typedef __attribute__((ext_vector_type(8)))  float    v8f;
typedef unsigned int u32x4 __attribute__((ext_vector_type(4)));
typedef int          i32x4 __attribute__((ext_vector_type(4)));
typedef int          i32x8 __attribute__((ext_vector_type(8)));

#define B_   56
#define T_   750
#define I_   188
#define IP   192          // I padded to multiple of 32
#define H_   512
#define G4   2048         // 4*H
#define MP   64           // batch rows padded to multiple of 16
#define KD   384000       // T_*H_
#define OUT_ 56
#define NKBLK  250        // split-K blocks for dense head
#define KCHUNK 1536       // K per dense block (250*1536 = 384000)

// LDS layout for the step kernel (bytes). 512-wide rows padded 1024B+16B
// (stride 1040 -> 4-dword bank rotation per row); 192-wide rows padded
// 16B per 128B (stride 432). All fragment reads stay 16B aligned.
#define ROWA    1040u
#define ROWX    432u
#define SA_OFF  0u                    // h slab      64 x 1040 = 66,560
#define SW_OFF  66560u                // W_hh slice  64 x 1040 = 66,560
#define SX_OFF  133120u               // x_t slab    64 x  432 = 27,648
#define SWI_OFF 160768u               // W_ih slice  64 x  432 = 27,648
#define SMEM_SZ 188416u               // total dynamic LDS

#if defined(__has_builtin)
#if __has_builtin(__builtin_amdgcn_tensor_load_to_lds)
#define HAVE_TDM 1
#endif
#endif
#ifndef HAVE_TDM
#define HAVE_TDM 0
#endif

union HFrag { v16h v; uint4 q[2]; };

__device__ inline v16h ldfrag2(const _Float16* p0, const _Float16* p1) {
  HFrag f;
  f.q[0] = *reinterpret_cast<const uint4*>(p0);
  f.q[1] = *reinterpret_cast<const uint4*>(p1);
  return f.v;
}

__device__ inline v16h zfrag() {
  HFrag f;
  f.q[0] = make_uint4(0u, 0u, 0u, 0u);
  f.q[1] = f.q[0];
  return f.v;
}

__device__ inline v8f zacc() {
  v8f z;
#pragma unroll
  for (int i = 0; i < 8; ++i) z[i] = 0.f;
  return z;
}

__device__ inline v8f wmma16(v16h a, v16h b, v8f c) {
  return __builtin_amdgcn_wmma_f32_16x16x32_f16(
      false, a, false, b, (short)0, c, false, false);
}

__device__ inline float sigm(float x) { return 1.f / (1.f + __expf(-x)); }
__device__ inline float tanh_f(float x) {
  float e = __expf(2.f * x);
  return 1.f - 2.f / (e + 1.f);
}

#if HAVE_TDM
// One TDM 2D tile load Global->LDS per CDNA5 D# layout (§8.3-8.5 of the
// async-tensor ISA chapter). tile==tensor dims; LDS-side row padding.
__device__ inline void tdm_load_2d(unsigned lds_off, const void* gp,
                                   unsigned tile0, unsigned tile1,
                                   unsigned stride0,          // elements
                                   unsigned padint, unsigned padamt) {
  unsigned long long ga = (unsigned long long)(uintptr_t)gp;
  u32x4 g0;
  g0[0] = 1u;                                        // count=1, user descriptor
  g0[1] = lds_off;                                   // lds_addr (bytes)
  g0[2] = (unsigned)(ga & 0xFFFFFFFFu);              // global_addr[31:0]
  g0[3] = (unsigned)((ga >> 32) & 0x01FFFFFFu)       // global_addr[56:32]
          | (2u << 30);                              // type = 2 ("image")
  i32x8 g1;
  g1[0] = (int)((1u << 16)                           // data_size = 2 bytes
                | (1u << 20)                         // pad_enable
                | (padint << 22)                     // pad_interval code
                | (padamt << 25));                   // pad_amount code
  g1[1] = (int)((tile0 & 0xFFFFu) << 16);            // tensor_dim0[15:0]
  g1[2] = (int)(((tile0 >> 16) & 0xFFFFu) | ((tile1 & 0xFFFFu) << 16));
  g1[3] = (int)(((tile1 >> 16) & 0xFFFFu) | ((tile0 & 0xFFFFu) << 16)); // tile_dim0
  g1[4] = (int)(tile1 & 0xFFFFu);                    // tile_dim1 (tile_dim2=0)
  g1[5] = (int)stride0;                              // tensor_dim0_stride[31:0]
  g1[6] = 0;
  g1[7] = 0;
  i32x4 z4 = {0, 0, 0, 0};
#if __clang_major__ >= 23
  i32x8 z8 = {0, 0, 0, 0, 0, 0, 0, 0};
  __builtin_amdgcn_tensor_load_to_lds(g0, g1, z4, z4, z8, 0);
#else
  __builtin_amdgcn_tensor_load_to_lds(g0, g1, z4, z4, 0);
#endif
}
#endif

// ---------------------------------------------------------------------------
// Conversion / init kernels
// ---------------------------------------------------------------------------
__global__ void conv_x_kernel(const float* __restrict__ x, _Float16* __restrict__ xh) {
  int idx = blockIdx.x * blockDim.x + threadIdx.x;
  if (idx >= MP * T_ * IP) return;
  int k = idx % IP, bt = idx / IP, t = bt % T_, b = bt / T_;
  float v = 0.f;
  if (b < B_ && k < I_) v = x[(b * T_ + t) * I_ + k];
  xh[idx] = (_Float16)v;
}

__global__ void conv_wih_kernel(const float* __restrict__ w, _Float16* __restrict__ wh) {
  int idx = blockIdx.x * blockDim.x + threadIdx.x;
  if (idx >= G4 * IP) return;
  int k = idx % IP, n = idx / IP;
  wh[idx] = (_Float16)((k < I_) ? w[n * I_ + k] : 0.f);
}

__global__ void conv_whh_kernel(const float* __restrict__ w, _Float16* __restrict__ wh) {
  int idx = blockIdx.x * blockDim.x + threadIdx.x;
  if (idx >= G4 * H_) return;
  wh[idx] = (_Float16)w[idx];
}

__global__ void bias_kernel(const float* __restrict__ bi, const float* __restrict__ bh,
                            float* __restrict__ bias) {
  int idx = blockIdx.x * blockDim.x + threadIdx.x;
  if (idx >= G4) return;
  bias[idx] = bi[idx] + bh[idx];
}

__global__ void zero_hc_kernel(_Float16* __restrict__ h, float* __restrict__ c) {
  int idx = blockIdx.x * blockDim.x + threadIdx.x;
  if (idx >= MP * H_) return;
  h[idx] = (_Float16)0.f;
  c[idx] = 0.f;
}

// ---------------------------------------------------------------------------
// Fragment bundles for the software-pipelined step loop.
// ---------------------------------------------------------------------------
struct StepFrags { v16h a, b0, b1, b2, b3; };

__device__ inline StepFrags ld_h_frags(const char* smem, int arow, int lrow,
                                       int half, int k0) {
  StepFrags f;
  const char* ap = smem + SA_OFF + arow * ROWA + 2 * k0 + 32 * half;
  f.a = ldfrag2((const _Float16*)ap, (const _Float16*)(ap + 16));
  const char* bb = smem + SW_OFF + lrow * ROWA + 2 * k0 + 32 * half;
  f.b0 = ldfrag2((const _Float16*)(bb),
                 (const _Float16*)(bb + 16));
  f.b1 = ldfrag2((const _Float16*)(bb + 16u * ROWA),
                 (const _Float16*)(bb + 16u * ROWA + 16));
  f.b2 = ldfrag2((const _Float16*)(bb + 32u * ROWA),
                 (const _Float16*)(bb + 32u * ROWA + 16));
  f.b3 = ldfrag2((const _Float16*)(bb + 48u * ROWA),
                 (const _Float16*)(bb + 48u * ROWA + 16));
  return f;
}

__device__ inline const char* xpad(const char* base, unsigned e) {
  return base + e + ((e >> 7) << 4);        // 16B pad per 128B interval
}

__device__ inline StepFrags ld_x_frags(const char* smem, int arow, int lrow,
                                       int half, int k0) {
  StepFrags f;
  unsigned e0 = arow * 384 + 2 * k0 + 32 * half;
  f.a = ldfrag2((const _Float16*)xpad(smem + SX_OFF, e0),
                (const _Float16*)xpad(smem + SX_OFF, e0 + 16));
#pragma unroll
  for (int g = 0; g < 4; ++g) {
    unsigned eb = (unsigned)(g * 16 + lrow) * 384 + 2 * k0 + 32 * half;
    const char* bp = xpad(smem + SWI_OFF, eb);
    (&f.b0)[g] = ldfrag2((const _Float16*)bp, (const _Float16*)(bp + 16));
  }
  return f;
}

// ---------------------------------------------------------------------------
// One LSTM time step. grid = 32 blocks (h-column groups) x 128 threads
// (4 waves = 4 m-tiles). gates = h_{t-1}@W_hh^T + x_t@W_ih^T + bias with the
// input projection fused (K = 512 + 192). Operands staged via TDM; inner
// loop register double-buffers the next k-step's 10 ds_loads ahead of the
// 4 WMMAs on the current fragments (continuous across the h->x boundary).
// ---------------------------------------------------------------------------
__global__ __launch_bounds__(128)
void lstm_step_kernel(const _Float16* __restrict__ xh,   // [64][750][192]
                      const _Float16* __restrict__ Wih,  // [2048][192]
                      const _Float16* __restrict__ Whh,  // [2048][512]
                      const float*    __restrict__ bias, // [2048]
                      _Float16*       __restrict__ h,    // [64][512] rows>=56 stay 0
                      float*          __restrict__ c,    // [64][512]
                      _Float16*       __restrict__ hs,   // [56][750][512]
                      int t)
{
  extern __shared__ char smem[];
  const int lane  = threadIdx.x & 31;
  const int wv    = threadIdx.x >> 5;     // m-tile 0..3
  const int hcol0 = blockIdx.x * 16;      // h column base
  const int half  = lane >> 4;
  const int lrow  = lane & 15;
  const int arow  = wv * 16 + lrow;       // padded batch row

#if HAVE_TDM
  if (threadIdx.x < 32) {                 // wave 0 issues all DMA descriptors
    tdm_load_2d(SA_OFF, h, 512, 64, 512, 7, 3);                 // h slab
    tdm_load_2d(SX_OFF, xh + t * IP, IP, 64, T_ * IP, 4, 3);    // x_t slab
#pragma unroll
    for (int g = 0; g < 4; ++g) {
      tdm_load_2d(SW_OFF + g * 16 * ROWA,
                  Whh + (g * H_ + hcol0) * H_, 512, 16, 512, 7, 3);
      tdm_load_2d(SWI_OFF + g * 16 * ROWX,
                  Wih + (g * H_ + hcol0) * IP, IP, 16, IP, 4, 3);
    }
    __builtin_amdgcn_s_wait_tensorcnt(0);
  }
#else
  // Cooperative staging fallback: identical padded LDS layout.
  for (int id = threadIdx.x; id < 64 * 64; id += 128) {   // h slab
    int r = id >> 6, s = id & 63;
    *(uint4*)(smem + SA_OFF + r * ROWA + s * 16) = *(const uint4*)(h + r * H_ + s * 8);
  }
  for (int id = threadIdx.x; id < 64 * 24; id += 128) {   // x slab
    int r = id / 24, s = id % 24;
    unsigned e = r * 384 + s * 16;
    *(uint4*)(smem + SX_OFF + e + ((e >> 7) << 4)) =
        *(const uint4*)(xh + (r * T_ + t) * IP + s * 8);
  }
  for (int id = threadIdx.x; id < 64 * 64; id += 128) {   // W_hh slice
    int r = id >> 6, s = id & 63, g = r >> 4, rr = r & 15;
    *(uint4*)(smem + SW_OFF + r * ROWA + s * 16) =
        *(const uint4*)(Whh + (g * H_ + hcol0 + rr) * H_ + s * 8);
  }
  for (int id = threadIdx.x; id < 64 * 24; id += 128) {   // W_ih slice
    int r = id / 24, s = id % 24, g = r >> 4, rr = r & 15;
    unsigned e = r * 384 + s * 16;
    *(uint4*)(smem + SWI_OFF + e + ((e >> 7) << 4)) =
        *(const uint4*)(Wih + (g * H_ + hcol0 + rr) * IP + s * 8);
  }
#endif
  if (t + 1 < T_) __builtin_prefetch(xh + (arow * T_ + t + 1) * IP, 0, 1);
  __syncthreads();

  v8f acc[4];
#pragma unroll
  for (int g = 0; g < 4; ++g) acc[g] = zacc();

  // ---- fused K loop, software pipelined: 16 h-steps then 6 x-steps ----
  StepFrags cur = ld_h_frags(smem, arow, lrow, half, 0);

#pragma unroll
  for (int k0 = 0; k0 < H_; k0 += 32) {
    StepFrags nxt = (k0 + 32 < H_)
                      ? ld_h_frags(smem, arow, lrow, half, k0 + 32)
                      : ld_x_frags(smem, arow, lrow, half, 0);
    acc[0] = wmma16(cur.a, cur.b0, acc[0]);
    acc[1] = wmma16(cur.a, cur.b1, acc[1]);
    acc[2] = wmma16(cur.a, cur.b2, acc[2]);
    acc[3] = wmma16(cur.a, cur.b3, acc[3]);
    cur = nxt;
  }
#pragma unroll
  for (int k0 = 0; k0 < IP; k0 += 32) {
    StepFrags nxt = cur;
    if (k0 + 32 < IP) nxt = ld_x_frags(smem, arow, lrow, half, k0 + 32);
    acc[0] = wmma16(cur.a, cur.b0, acc[0]);
    acc[1] = wmma16(cur.a, cur.b1, acc[1]);
    acc[2] = wmma16(cur.a, cur.b2, acc[2]);
    acc[3] = wmma16(cur.a, cur.b3, acc[3]);
    cur = nxt;
  }

  // ---- gate nonlinearities + state update ----
  // C layout: lane l holds col n=l%16; VGPR v holds row m = v + 8*(l/16)
  const int n  = hcol0 + lrow;
  const float b0 = bias[0 * H_ + n];
  const float b1 = bias[1 * H_ + n];
  const float b2 = bias[2 * H_ + n];
  const float b3 = bias[3 * H_ + n];

#pragma unroll
  for (int v = 0; v < 8; ++v) {
    const int brow = wv * 16 + v + half * 8;
    if (brow < B_) {
      float si = sigm(acc[0][v] + b0);
      float sf = sigm(acc[1][v] + b1);
      float tg = tanh_f(acc[2][v] + b2);
      float so = sigm(acc[3][v] + b3);
      float cn = sf * c[brow * H_ + n] + si * tg;
      float hn = so * tanh_f(cn);
      c[brow * H_ + n] = cn;
      _Float16 hh = (_Float16)hn;
      h[brow * H_ + n] = hh;
      hs[(brow * T_ + t) * H_ + n] = hh;
    }
  }
}

// ---------------------------------------------------------------------------
// Dense head, split-K: 250 blocks x 128 threads; W_dense staged fp32->f16
// through double-buffered LDS (one barrier per k-step, staging overlapped
// with WMMA); deterministic partials.
// ---------------------------------------------------------------------------
__global__ __launch_bounds__(128)
void dense_kernel(const _Float16* __restrict__ hs,   // [56][384000]
                  const float*    __restrict__ Wd,   // [56][384000]
                  float*          __restrict__ part) // [250][64][64]
{
  __shared__ _Float16 sB[2][64 * 32];

  const int lane = threadIdx.x & 31;
  const int wv   = threadIdx.x >> 5;
  const int half = lane >> 4;
  const int lrow = lane & 15;
  const int kbase = blockIdx.x * KCHUNK;

  v8f acc[4];
#pragma unroll
  for (int nt = 0; nt < 4; ++nt) acc[nt] = zacc();

  const int arow = wv * 16 + lrow;
  const _Float16* hrow = hs + arow * KD;

  const int sn = threadIdx.x >> 1;          // staged W_dense row (0..63)
  const int sk = (threadIdx.x & 1) * 16;    // staged k half

  // stage first tile
  {
    const int k0 = kbase;
#pragma unroll
    for (int j = 0; j < 16; j += 4) {
      float4 wq = make_float4(0.f, 0.f, 0.f, 0.f);
      if (sn < B_) wq = *reinterpret_cast<const float4*>(Wd + sn * KD + k0 + sk + j);
      sB[0][sn * 32 + sk + j + 0] = (_Float16)wq.x;
      sB[0][sn * 32 + sk + j + 1] = (_Float16)wq.y;
      sB[0][sn * 32 + sk + j + 2] = (_Float16)wq.z;
      sB[0][sn * 32 + sk + j + 3] = (_Float16)wq.w;
    }
  }
  __syncthreads();

  int buf = 0;
  for (int ks = 0; ks < KCHUNK; ks += 32) {
    const int k0 = kbase + ks;

    // compute from sB[buf]
    v16h a = (arow < B_)
               ? ldfrag2(hrow + k0 + half * 16, hrow + k0 + 8 + half * 16)
               : zfrag();
#pragma unroll
    for (int nt = 0; nt < 4; ++nt) {
      const _Float16* bp = &sB[buf][(nt * 16 + lrow) * 32 + half * 16];
      v16h b = ldfrag2(bp, bp + 8);
      acc[nt] = wmma16(a, b, acc[nt]);
    }

    // stage next tile into sB[buf^1] (overlaps with this step's WMMAs)
    if (ks + 32 < KCHUNK) {
      const int kn = k0 + 32;
#pragma unroll
      for (int j = 0; j < 16; j += 4) {
        float4 wq = make_float4(0.f, 0.f, 0.f, 0.f);
        if (sn < B_) wq = *reinterpret_cast<const float4*>(Wd + sn * KD + kn + sk + j);
        sB[buf ^ 1][sn * 32 + sk + j + 0] = (_Float16)wq.x;
        sB[buf ^ 1][sn * 32 + sk + j + 1] = (_Float16)wq.y;
        sB[buf ^ 1][sn * 32 + sk + j + 2] = (_Float16)wq.z;
        sB[buf ^ 1][sn * 32 + sk + j + 3] = (_Float16)wq.w;
      }
    }
    __syncthreads();
    buf ^= 1;
  }

  float* pb = part + (size_t)blockIdx.x * 4096;
#pragma unroll
  for (int nt = 0; nt < 4; ++nt) {
#pragma unroll
    for (int v = 0; v < 8; ++v) {
      const int m = wv * 16 + v + half * 8;
      const int n = nt * 16 + lrow;
      pb[m * 64 + n] = acc[nt][v];
    }
  }
}

__global__ void reduce_kernel(const float* __restrict__ part,
                              const float* __restrict__ bd,
                              float* __restrict__ out) {
  int idx = blockIdx.x * blockDim.x + threadIdx.x;
  if (idx >= B_ * OUT_) return;
  int b = idx / OUT_, o = idx % OUT_;
  float s = 0.f;
  for (int blk = 0; blk < NKBLK; ++blk)
    s += part[blk * 4096 + b * 64 + o];
  out[idx] = s + bd[o];
}

// ---------------------------------------------------------------------------
// Host side
// ---------------------------------------------------------------------------
extern "C" void kernel_launch(void* const* d_in, const int* in_sizes, int n_in,
                              void* d_out, int out_size, void* d_ws, size_t ws_size,
                              hipStream_t stream) {
  const float* x   = (const float*)d_in[0];
  const float* Wih = (const float*)d_in[1];
  const float* Whh = (const float*)d_in[2];
  const float* bih = (const float*)d_in[3];
  const float* bhh = (const float*)d_in[4];
  const float* Wd  = (const float*)d_in[5];
  const float* bd  = (const float*)d_in[6];
  float* out = (float*)d_out;

  char* w = (char*)d_ws;
  _Float16* xh   = (_Float16*)(w);               // 18,432,000
  _Float16* WihH = (_Float16*)(w + 18432000);    //    786,432
  _Float16* WhhH = (_Float16*)(w + 19218432);    //  2,097,152
  float*    bias = (float*)   (w + 21315584);    //      8,192
  _Float16* h    = (_Float16*)(w + 21323776);    //     65,536
  float*    c    = (float*)   (w + 21389312);    //    131,072
  _Float16* hs   = (_Float16*)(w + 21520384);    // 43,008,000
  float*    part = (float*)   (w + 64528384);    //  4,096,000
  (void)ws_size; (void)in_sizes; (void)n_in; (void)out_size;

  conv_x_kernel  <<<(MP * T_ * IP + 255) / 256, 256, 0, stream>>>(x, xh);
  conv_wih_kernel<<<(G4 * IP + 255) / 256,      256, 0, stream>>>(Wih, WihH);
  conv_whh_kernel<<<(G4 * H_ + 255) / 256,      256, 0, stream>>>(Whh, WhhH);
  bias_kernel    <<<(G4 + 255) / 256,           256, 0, stream>>>(bih, bhh, bias);
  zero_hc_kernel <<<(MP * H_ + 255) / 256,      256, 0, stream>>>(h, c);

  for (int t = 0; t < T_; ++t) {
    lstm_step_kernel<<<32, 128, SMEM_SZ, stream>>>(xh, WihH, WhhH, bias, h, c, hs, t);
  }

  dense_kernel <<<NKBLK, 128, 0, stream>>>(hs, Wd, part);
  reduce_kernel<<<(B_ * OUT_ + 255) / 256, 256, 0, stream>>>(part, bd, out);
}
